// AttnMessageAggregator_72052371357802
// MI455X (gfx1250) — matrix operand order repaired and checked by reading, not compiled, not run on hardware
//
#include <hip/hip_runtime.h>
#include <math.h>

// ---------------------------------------------------------------------------
// AttnMessageAggregator for MI455X (gfx1250, wave32, WMMA).
//
// Roofline: mandatory bytes = messages 167.8MB + out 16.8MB ~= 185MB
//   -> ~8us @ 23.3 TB/s.  Algebraically reduced FLOPs ~= 9 GFLOP (only the
//   last query row is needed; K/V projections are folded through the tiny
//   L=10 attention).  f16 WMMA (f32 accumulate) keeps matrix work far below
//   the tensor ceiling -> kernel is HBM-streaming bound.
//
// This revision software-pipelines the GEMM inner loops: B fragments for
// N-tile nt+1 are prefetched (double-buffered) before the WMMA chain of
// tile nt, so the wait before each WMMA chain drains only to the prefetch
// depth instead of s_wait_loadcnt 0 per K-step.
// ---------------------------------------------------------------------------

typedef _Float16 h16;
typedef __attribute__((ext_vector_type(16))) _Float16 v16h;
typedef __attribute__((ext_vector_type(8)))  _Float16 v8h;
typedef __attribute__((ext_vector_type(2)))  _Float16 h2v;
typedef __attribute__((ext_vector_type(8)))  float    v8f;
typedef __attribute__((ext_vector_type(4)))  float    vf4;
typedef __attribute__((ext_vector_type(2)))  float    vf2;

#define NSEQ 16384
#define LSEQ 10
#define DIM  256
#define DHH  128   // head dim

// ---- fragment builders (CDNA5 WMMA VGPR layouts, wave32) -------------------
// A (16x32 f16): lane m=l&15; lanes16-31 shift K by 8; VGPR i: K=2i,2i+1 (+16 for i>=4)
__device__ __forceinline__ v16h afrag_f16(const h16* __restrict__ row, int kc, int lane) {
  const int kh = (lane >> 4) << 3;
  v16h a;
#pragma unroll
  for (int i = 0; i < 4; ++i) {
    h2v p0 = *(const h2v*)(row + kc + kh + 2 * i);
    h2v p1 = *(const h2v*)(row + kc + 16 + kh + 2 * i);
    a[2*i]   = p0[0]; a[2*i+1] = p0[1];
    a[8+2*i] = p1[0]; a[9+2*i] = p1[1];
  }
  return a;
}

__device__ __forceinline__ v16h afrag_f32(const float* __restrict__ row, int kc, int lane) {
  const int kh = (lane >> 4) << 3;
  v16h a;
#pragma unroll
  for (int i = 0; i < 4; ++i) {
    vf2 p0 = *(const vf2*)(row + kc + kh + 2 * i);
    vf2 p1 = *(const vf2*)(row + kc + 16 + kh + 2 * i);
    a[2*i]   = (h16)p0[0]; a[2*i+1] = (h16)p0[1];
    a[8+2*i] = (h16)p1[0]; a[9+2*i] = (h16)p1[1];
  }
  return a;
}

// B (32x16 f16): lane n=l&15; lanes16-31 shift K by 8; VGPR i: K=2i,2i+1 (+16 for i>=4)
// 'col' = start of the K-contiguous strip for this lane's column.
__device__ __forceinline__ v16h bfrag_f16(const h16* __restrict__ col, int kc, int lane) {
  const int kb = (lane >> 4) << 3;
  v16h b;
#pragma unroll
  for (int i = 0; i < 8; ++i) {
    h2v p = *(const h2v*)(col + kc + kb + (i >> 2) * 8 + 2 * i);
    b[2*i] = p[0]; b[2*i+1] = p[1];
  }
  return b;
}

// ---- software-pipelined GEMM over N-tiles ----------------------------------
// C(16 x NT*16) = A(16 x KC*32) @ B, B stored K-contiguous per output column:
// element B[k][n] at B0[(nt*16+n)*DIM + k].  Double-buffers B fragments so the
// loads for tile nt+1 are in flight during the WMMA chain of tile nt.
template <int NT, int KC, typename EmitFn>
__device__ __forceinline__ void wmma_gemm(const v16h* __restrict__ am,
                                          const h16* __restrict__ B0,
                                          int ncol, int lane, EmitFn&& emit) {
  v16h buf[2][KC];
#pragma unroll
  for (int kc = 0; kc < KC; ++kc)
    buf[0][kc] = bfrag_f16(B0 + (size_t)ncol * DIM, kc * 32, lane);
#pragma unroll
  for (int nt = 0; nt < NT; ++nt) {
    if (nt + 1 < NT) {
      const h16* bcol = B0 + (size_t)((nt + 1) * 16 + ncol) * DIM;
#pragma unroll
      for (int kc = 0; kc < KC; ++kc)
        buf[(nt + 1) & 1][kc] = bfrag_f16(bcol, kc * 32, lane);
    }
    v8f acc = {};
#pragma unroll
    for (int kc = 0; kc < KC; ++kc)
      acc = __builtin_amdgcn_wmma_f32_16x16x32_f16(false, am[kc], false, buf[nt & 1][kc],
                                                   (short)0, acc, false, false);
    emit(nt, acc);
  }
}

// ---- weight prep: f32 -> f16, wk transposed so all B reads are K-contiguous
__global__ void prep_weights_kernel(const float* __restrict__ w_in,
                                    const float* __restrict__ w_out,
                                    h16* __restrict__ WQ,  h16* __restrict__ WKT,
                                    h16* __restrict__ WV,  h16* __restrict__ WOUT) {
  int gid = blockIdx.x * blockDim.x + threadIdx.x;   // 0 .. 4*65536-1
  int sec = gid >> 16;
  int i   = gid & 65535;
  if (sec == 0)      WQ[i]   = (h16)w_in[i];                         // wq row-major
  else if (sec == 1) { int d = i & 255, e = i >> 8;
                       WKT[i] = (h16)w_in[65536 + d * 256 + e]; }    // wkT[e][d]
  else if (sec == 2) WV[i]   = (h16)w_in[131072 + i];                // wv row-major
  else               WOUT[i] = (h16)w_out[i];                        // wout row-major
}

// ---- fused main kernel: 1 wave per 16 sequences ----------------------------
__global__ __launch_bounds__(32)
void attn_agg_kernel(const float* __restrict__ msgs, const float* __restrict__ ts,
                     const float* __restrict__ w_time, const float* __restrict__ b_time,
                     const float* __restrict__ b_in,  const float* __restrict__ b_out,
                     const int*   __restrict__ lengths,
                     const h16* __restrict__ WQ,  const h16* __restrict__ WKT,
                     const h16* __restrict__ WV,  const h16* __restrict__ WOUT,
                     float* __restrict__ out) {
  // phase-overlapped LDS: [0,32768) = R(f32 16x512) then {Ah(2x16x256 h16), CTX}
  //                       [32768,40960) = Q (16x256 h16), [40960,..) = attn wts
  __shared__ __align__(16) char smem[43520];
  float* Rlds = (float*)smem;
  h16*   Ah   = (h16*)smem;                 // 2 x 16x256 f16 (phase 4/5)
  h16*   CTX  = (h16*)(smem + 16384);       // 16x256 f16
  h16*   Q    = (h16*)(smem + 32768);       // 16x256 f16
  float* Awt  = (float*)(smem + 40960);     // 16 x (2*10) attention weights

  const int lane = threadIdx.x;
  const int seq0 = blockIdx.x * 16;
  const int mrow = lane & 15;
  const int mb   = (lane >> 4) * 8;         // C-frag row base
  const int ncol = lane & 15;               // C/B-frag column

  // ---- Phase 1: Q = messages[:, L-1, :] @ wq^T + bq  (16x256) --------------
  {
    const float* arow = msgs + (size_t)(seq0 + mrow) * (LSEQ * DIM) + (LSEQ - 1) * DIM;
    v16h am[8];
#pragma unroll
    for (int kc = 0; kc < 8; ++kc) am[kc] = afrag_f32(arow, kc * 32, lane);
    wmma_gemm<16, 8>(am, WQ, ncol, lane, [&](int nt, v8f acc) {
      float bias = b_in[nt * 16 + ncol];
#pragma unroll
      for (int i = 0; i < 8; ++i)
        Q[(mb + i) * DIM + nt * 16 + ncol] = (h16)(acc[i] + bias);
    });
  }
  __syncthreads();

  // ---- Phase 2: R_h = Q_h (16x128) @ wk_h (128x256)  -> Rlds (16x512 f32) --
#pragma unroll
  for (int h = 0; h < 2; ++h) {
    v16h am[4];
#pragma unroll
    for (int kc = 0; kc < 4; ++kc)
      am[kc] = afrag_f16(Q + mrow * DIM + h * DHH, kc * 32, lane);
    wmma_gemm<16, 4>(am, WKT + h * DHH, ncol, lane, [&](int nt, v8f acc) {
#pragma unroll
      for (int i = 0; i < 8; ++i)
        Rlds[(mb + i) * 512 + h * DIM + nt * 16 + ncol] = acc[i];
    });
  }
  __syncthreads();

  // ---- Phase 3: scores = (R . cos(dt*w+b) + q.bk)/sqrt(128), mask, softmax -
  {
    const int s = lane & 15;
    const int p = lane >> 4;                // e-half split across lane halves
    const float* tp = ts + (size_t)(seq0 + s) * LSEQ;
    float t9 = tp[LSEQ - 1];
    float dl[LSEQ];
#pragma unroll
    for (int j = 0; j < LSEQ; ++j) dl[j] = tp[j] - t9;

    float sq0 = 0.f, sq1 = 0.f;             // q_h . bk_h
    for (int dh = p * 64; dh < p * 64 + 64; dh += 2) {
      h2v q0 = *(const h2v*)(Q + s * DIM + dh);
      h2v q1 = *(const h2v*)(Q + s * DIM + DHH + dh);
      vf2 k0 = *(const vf2*)(b_in + DIM + dh);
      vf2 k1 = *(const vf2*)(b_in + DIM + DHH + dh);
      sq0 += (float)q0[0] * k0[0] + (float)q0[1] * k0[1];
      sq1 += (float)q1[0] * k1[0] + (float)q1[1] * k1[1];
    }
    float sc0[LSEQ] = {}, sc1[LSEQ] = {};
    for (int e = p * 128; e < p * 128 + 128; ++e) {
      float r0 = Rlds[s * 512 + e];
      float r1 = Rlds[s * 512 + DIM + e];
      float w  = w_time[e], bt = b_time[e];
#pragma unroll
      for (int j = 0; j < LSEQ; ++j) {
        float t = cosf(fmaf(dl[j], w, bt));
        sc0[j] = fmaf(r0, t, sc0[j]);
        sc1[j] = fmaf(r1, t, sc1[j]);
      }
    }
    sq0 += __shfl_xor(sq0, 16, 32);
    sq1 += __shfl_xor(sq1, 16, 32);
#pragma unroll
    for (int j = 0; j < LSEQ; ++j) {
      sc0[j] += __shfl_xor(sc0[j], 16, 32);
      sc1[j] += __shfl_xor(sc1[j], 16, 32);
    }
    const float scale = 0.08838834764831845f;   // 1/sqrt(128)
    int jmin = (LSEQ - 1) - lengths[seq0 + s];
#pragma unroll
    for (int j = 0; j < LSEQ; ++j) {
      sc0[j] = (j < jmin) ? -1.0e9f : (sc0[j] + sq0) * scale;
      sc1[j] = (j < jmin) ? -1.0e9f : (sc1[j] + sq1) * scale;
    }
    if (lane < 16) {
      float m0 = sc0[0], m1 = sc1[0];
#pragma unroll
      for (int j = 1; j < LSEQ; ++j) { m0 = fmaxf(m0, sc0[j]); m1 = fmaxf(m1, sc1[j]); }
      float z0 = 0.f, z1 = 0.f;
#pragma unroll
      for (int j = 0; j < LSEQ; ++j) {
        sc0[j] = expf(sc0[j] - m0); z0 += sc0[j];
        sc1[j] = expf(sc1[j] - m1); z1 += sc1[j];
      }
      float iz0 = 1.f / z0, iz1 = 1.f / z1;
#pragma unroll
      for (int j = 0; j < LSEQ; ++j) {
        Awt[s * 20 + j]      = sc0[j] * iz0;
        Awt[s * 20 + 10 + j] = sc1[j] * iz1;
      }
    }
  }
  __syncthreads();

  // ---- Phase 4 (HBM-bound): m_h = sum_j a[h,j]*messages[j]  -> Ah f16 ------
  {
    const int e0 = lane * 8;
    for (int s = 0; s < 16; ++s) {
      const float* mp = msgs + (size_t)(seq0 + s) * (LSEQ * DIM) + e0;
      float a0acc[8] = {}, a1acc[8] = {};
#pragma unroll
      for (int j = 0; j < LSEQ; ++j) {
        vf4 x0 = *(const vf4*)(mp + j * DIM);
        vf4 x1 = *(const vf4*)(mp + j * DIM + 4);
        float a0 = Awt[s * 20 + j];
        float a1 = Awt[s * 20 + 10 + j];
#pragma unroll
        for (int t = 0; t < 4; ++t) {
          a0acc[t]     = fmaf(a0, x0[t], a0acc[t]);
          a0acc[4 + t] = fmaf(a0, x1[t], a0acc[4 + t]);
          a1acc[t]     = fmaf(a1, x0[t], a1acc[t]);
          a1acc[4 + t] = fmaf(a1, x1[t], a1acc[4 + t]);
        }
        if (s < 15) __builtin_prefetch(mp + LSEQ * DIM + j * DIM, 0, 1);
      }
      v8h m0, m1;
#pragma unroll
      for (int t = 0; t < 8; ++t) { m0[t] = (h16)a0acc[t]; m1[t] = (h16)a1acc[t]; }
      *(v8h*)(Ah + s * DIM + e0)        = m0;
      *(v8h*)(Ah + 4096 + s * DIM + e0) = m1;
    }
  }
  __syncthreads();

  // ---- Phase 5: ctx_h = m_h (16x256) @ wv_h^T (256x128) + bv_h  -> CTX -----
#pragma unroll
  for (int h = 0; h < 2; ++h) {
    const h16* arow = Ah + h * 4096 + mrow * DIM;
    v16h am[8];
#pragma unroll
    for (int kc = 0; kc < 8; ++kc) am[kc] = afrag_f16(arow, kc * 32, lane);
    wmma_gemm<8, 8>(am, WV + (size_t)h * DHH * DIM, ncol, lane, [&](int nt, v8f acc) {
      float bias = b_in[2 * DIM + h * DHH + nt * 16 + ncol];
#pragma unroll
      for (int i = 0; i < 8; ++i)
        CTX[(mb + i) * DIM + h * DHH + nt * 16 + ncol] = (h16)(acc[i] + bias);
    });
  }
  __syncthreads();

  // ---- Phase 6: out = ctx (16x256) @ wout^T (256x256) + b_out --------------
  {
    const h16* arow = CTX + mrow * DIM;
    v16h am[8];
#pragma unroll
    for (int kc = 0; kc < 8; ++kc) am[kc] = afrag_f16(arow, kc * 32, lane);
    wmma_gemm<16, 8>(am, WOUT, ncol, lane, [&](int nt, v8f acc) {
      float bias = b_out[nt * 16 + ncol];
#pragma unroll
      for (int i = 0; i < 8; ++i)
        out[(size_t)(seq0 + mb + i) * DIM + nt * 16 + ncol] = acc[i] + bias;
    });
  }
}

extern "C" void kernel_launch(void* const* d_in, const int* in_sizes, int n_in,
                              void* d_out, int out_size, void* d_ws, size_t ws_size,
                              hipStream_t stream) {
  const float* msgs    = (const float*)d_in[0];
  const float* ts      = (const float*)d_in[1];
  const float* w_time  = (const float*)d_in[2];
  const float* b_time  = (const float*)d_in[3];
  const float* w_in    = (const float*)d_in[4];
  const float* b_in    = (const float*)d_in[5];
  const float* w_out   = (const float*)d_in[6];
  const float* b_out   = (const float*)d_in[7];
  const int*   lengths = (const int*)d_in[8];
  float* out = (float*)d_out;

  h16* WQ   = (h16*)d_ws;          // 65536 f16 each (512 KB total in d_ws)
  h16* WKT  = WQ  + 65536;
  h16* WV   = WKT + 65536;
  h16* WOUT = WV  + 65536;

  prep_weights_kernel<<<1024, 256, 0, stream>>>(w_in, w_out, WQ, WKT, WV, WOUT);
  attn_agg_kernel<<<NSEQ / 16, 32, 0, stream>>>(msgs, ts, w_time, b_time, b_in, b_out,
                                                lengths, WQ, WKT, WV, WOUT, out);
}
